// Decoder_81217831567949
// MI455X (gfx1250) — compile-verified, matrix operand order checked by reading, not run on hardware
//
#include <hip/hip_runtime.h>
#include <math.h>

// ---------------- CDNA5 WMMA / TDM types ----------------
typedef __attribute__((ext_vector_type(16))) __bf16          v16bf;
typedef __attribute__((ext_vector_type(16))) unsigned short  v16us;
typedef __attribute__((ext_vector_type(8)))  unsigned short  v8us;
typedef __attribute__((ext_vector_type(8)))  float           v8f;
typedef __attribute__((ext_vector_type(4)))  unsigned int    v4u;
typedef __attribute__((ext_vector_type(8)))  int             v8i_;
typedef __attribute__((ext_vector_type(4)))  int             v4i_;

#define S_LEN   1024
#define DMODEL  4096
#define NHEAD   32
#define NKV     8
#define HD      128
#define TOPK    128

__device__ __forceinline__ unsigned short f2bf(float f) {
  unsigned int b = __builtin_bit_cast(unsigned int, f);
  b += 0x7FFFu + ((b >> 16) & 1u);          // round-to-nearest-even
  return (unsigned short)(b >> 16);
}

__device__ __forceinline__ v16bf ld16(const unsigned short* __restrict__ p) {
  v16us u = *(const v16us*)p;               // 32B-aligned
  return __builtin_bit_cast(v16bf, u);
}

__device__ __forceinline__ v16bf ld16_2(const unsigned short* p) {  // 2x16B (padded LDS)
  v8us a = *(const v8us*)p;
  v8us b = *(const v8us*)(p + 8);
  v16us u;
#pragma unroll
  for (int i = 0; i < 8; ++i) { u[i] = a[i]; u[8 + i] = b[i]; }
  return __builtin_bit_cast(v16bf, u);
}

__device__ __forceinline__ v8f wmma_bf16(v16bf a, v16bf b, v8f c) {
  return __builtin_amdgcn_wmma_f32_16x16x32_bf16(false, a, false, b,
                                                 (short)0, c, false, false);
}

// ---------------------------------------------------------------------------
// TDM: DMA a 128-row x 32-element (bf16) tile from global to LDS, with LDS
// row padding 64B data + 16B pad (pitch 80B). D# per CDNA5 ISA 8.3/8.4.
// Toolchain uses the 6-arg builtin: (u32x4 g0, i32x8 g1, i32x4, i32x4, i32x8, cpol)
// ---------------------------------------------------------------------------
__device__ __forceinline__ void tdm_load_tile(const unsigned short* lds_p,
                                              const unsigned short* g_p,
                                              unsigned stride_elems) {
  unsigned lds_off = (unsigned)(size_t)lds_p;          // low 32b = LDS byte offset
  unsigned long long ga = (unsigned long long)(size_t)g_p;
  v4u g0;
  g0[0] = 1u;                                          // count=1, user mode
  g0[1] = lds_off;                                     // lds_addr
  g0[2] = (unsigned)ga;                                // global_addr[31:0]
  g0[3] = ((unsigned)(ga >> 32) & 0x01FFFFFFu) | 0x80000000u;  // addr[56:32] | type=2
  v8i_ g1;
  // data_size=1 (2B), pad_enable, pad_interval=3 (16 DW = 64B), pad_amount=3 (4 DW = 16B)
  g1[0] = (int)((1u << 16) | (1u << 20) | (3u << 22) | (3u << 25));
  g1[1] = (int)(32u << 16);      // tensor_dim0 = 32  (lo16 at bits 63:48)
  g1[2] = (int)(128u << 16);     // tensor_dim0 hi = 0 ; tensor_dim1 = 128 (lo16)
  g1[3] = (int)(32u << 16);      // tensor_dim1 hi = 0 ; tile_dim0 = 32
  g1[4] = (int)128u;             // tile_dim1 = 128 ; tile_dim2 = 0
  g1[5] = (int)stride_elems;     // tensor_dim0_stride[31:0] (= K)
  g1[6] = 0;                     // stride hi / dim1_stride lo
  g1[7] = 0;
  v4i_ z4 = {};
  v8i_ z8 = {};
  __builtin_amdgcn_tensor_load_to_lds(g0, g1, z4, z4, z8, 0);
}

// ---------------------------------------------------------------------------
// f32 -> bf16 bulk conversion (bandwidth-bound; once per tensor)
// ---------------------------------------------------------------------------
__global__ __launch_bounds__(256)
void cvt_bf16(const float* __restrict__ in, unsigned short* __restrict__ out, int n) {
  int i = (blockIdx.x * blockDim.x + threadIdx.x) * 8;
  if (i + 8 <= n) {
    float4 a = *(const float4*)(in + i);
    float4 b = *(const float4*)(in + i + 4);
    v8us u;
    u[0] = f2bf(a.x); u[1] = f2bf(a.y); u[2] = f2bf(a.z); u[3] = f2bf(a.w);
    u[4] = f2bf(b.x); u[5] = f2bf(b.y); u[6] = f2bf(b.z); u[7] = f2bf(b.w);
    *(v8us*)(out + i) = u;
  }
}

// ---------------------------------------------------------------------------
// Pure-bf16 GEMM: C[M,N] = A[M,K] @ B[N,K]^T, f32 accumulate/out.
// 256 threads (8 waves); tile 128M x 128N; B tile DMA'd by the Tensor Data
// Mover into double-buffered padded LDS, prefetch overlapped with WMMA.
// ---------------------------------------------------------------------------
__device__ __forceinline__ void gemm_step(const unsigned short (*Bsb)[40],
                                          const unsigned short* arow, int k0,
                                          v8f* acc, int l16, int half) {
  v16bf af = ld16(arow + k0);
  v16bf bfr[8];
#pragma unroll
  for (int nt = 0; nt < 8; ++nt) bfr[nt] = ld16_2(&Bsb[nt * 16 + l16][half * 16]);
#pragma unroll
  for (int nt = 0; nt < 8; ++nt) acc[nt] = wmma_bf16(af, bfr[nt], acc[nt]);
}

__global__ __launch_bounds__(256)
void gemm_bb(const unsigned short* __restrict__ A,
             const unsigned short* __restrict__ B,
             float* __restrict__ C, int M, int N, int K) {
  __shared__ unsigned short Bs[2][128][40];   // 2 x 10KB, 80B row pitch
  const int lane = threadIdx.x & 31;
  const int wave = threadIdx.x >> 5;
  const int half = lane >> 4;
  const int l16  = lane & 15;
  const int m0 = blockIdx.y * 128 + wave * 16;
  const int n0 = blockIdx.x * 128;

  v8f zero = {};
  v8f acc[8];
#pragma unroll
  for (int i = 0; i < 8; ++i) acc[i] = zero;

  const unsigned short* arow  = A + (size_t)(m0 + l16) * K + half * 16;
  const unsigned short* btile = B + (size_t)n0 * K;     // 128 rows, stride K

  if (wave == 0) tdm_load_tile(&Bs[0][0][0], btile, (unsigned)K);

  for (int k0 = 0; k0 < K; k0 += 64) {
    if (wave == 0) __builtin_amdgcn_s_wait_tensorcnt(0);
    __syncthreads();
    if (wave == 0) tdm_load_tile(&Bs[1][0][0], btile + k0 + 32, (unsigned)K);
    gemm_step(Bs[0], arow, k0, acc, l16, half);

    if (wave == 0) __builtin_amdgcn_s_wait_tensorcnt(0);
    __syncthreads();
    if (wave == 0 && k0 + 64 < K) tdm_load_tile(&Bs[0][0][0], btile + k0 + 64, (unsigned)K);
    gemm_step(Bs[1], arow, k0 + 32, acc, l16, half);
  }
  // C/D layout: VGPR j -> (m = j + 8*half, n = l16)
#pragma unroll
  for (int nt = 0; nt < 8; ++nt)
#pragma unroll
    for (int j = 0; j < 8; ++j)
      C[(size_t)(m0 + half * 8 + j) * N + n0 + nt * 16 + l16] = acc[nt][j];
}

// ---------------------------------------------------------------------------
// RoPE + pack: qr[H][S][HD], kr[NKV][S][HD], vT[NKV][HD][S]  (all bf16)
// ---------------------------------------------------------------------------
__global__ __launch_bounds__(256)
void rope_pack(const float* __restrict__ Qf, const float* __restrict__ Kf,
               const float* __restrict__ Vf,
               unsigned short* __restrict__ qr, unsigned short* __restrict__ kr,
               unsigned short* __restrict__ vT) {
  const float LOG_THETA = 9.210340371976184f;  // ln(10000)
  int idx = blockIdx.x * blockDim.x + threadIdx.x;
  if (idx < S_LEN * DMODEL) {                       // ---- Q ----
    int s = idx / DMODEL, c = idx % DMODEL;
    int d = c & (HD - 1), i = d & 63, h = c >> 7;
    float ang = (float)s * __expf(-LOG_THETA * (float)i / 64.0f);
    float x  = Qf[idx];
    float xr = (d < 64) ? -Qf[idx + 64] : Qf[idx - 64];
    qr[((size_t)h * S_LEN + s) * HD + d] = f2bf(x * __cosf(ang) + xr * __sinf(ang));
  } else if (idx < S_LEN * DMODEL + S_LEN * NKV * HD) {  // ---- K ----
    int j = idx - S_LEN * DMODEL;
    int s = j / (NKV * HD), c = j % (NKV * HD);
    int d = c & (HD - 1), i = d & 63, h = c >> 7;
    float ang = (float)s * __expf(-LOG_THETA * (float)i / 64.0f);
    float x  = Kf[j];
    float xr = (d < 64) ? -Kf[j + 64] : Kf[j - 64];
    kr[((size_t)h * S_LEN + s) * HD + d] = f2bf(x * __cosf(ang) + xr * __sinf(ang));
  } else {                                               // ---- V transpose ----
    int j = idx - S_LEN * DMODEL - S_LEN * NKV * HD;
    if (j < S_LEN * NKV * HD) {
      int s = j / (NKV * HD), c = j % (NKV * HD);
      int d = c & (HD - 1), h = c >> 7;
      vT[((size_t)h * HD + d) * S_LEN + s] = f2bf(Vf[j]);
    }
  }
}

// ---------------------------------------------------------------------------
// Sparse attention: one block per (head, 16-query tile); 128 threads (4 waves).
// ---------------------------------------------------------------------------
__global__ __launch_bounds__(128)
void attn_sparse(const unsigned short* __restrict__ qr,
                 const unsigned short* __restrict__ kr,
                 const unsigned short* __restrict__ vT,
                 unsigned short* __restrict__ outb /* [S][DMODEL] bf16 */) {
  __shared__ float sc[16][S_LEN];   // 64 KB
  const int lane = threadIdx.x & 31;
  const int wave = threadIdx.x >> 5;
  const int half = lane >> 4;
  const int l16  = lane & 15;
  const int h    = blockIdx.y;
  const int kvh  = h >> 2;          // GQA repeat = 4
  const int q0   = blockIdx.x * 16;

  v8f zero = {};
  const float scale = 0.08838834764831845f;   // 1/sqrt(128)
  const float NEG   = -3.4028235e38f;         // finfo(f32).min

  // ---- phase 1: scores ----
  const unsigned short* qbase = qr + ((size_t)h * S_LEN + q0 + l16) * HD + half * 16;
  v16bf aq[4];
#pragma unroll
  for (int t = 0; t < 4; ++t) aq[t] = ld16(qbase + t * 32);

  for (int kt = wave; kt < S_LEN / 16; kt += 4) {
    const int kbase = kt * 16;
    const unsigned short* kb =
        kr + ((size_t)kvh * S_LEN + kbase + l16) * HD + half * 16;
    v16bf bk[4];
#pragma unroll
    for (int t = 0; t < 4; ++t) bk[t] = ld16(kb + t * 32);
    v8f c = zero;
#pragma unroll
    for (int t = 0; t < 4; ++t) c = wmma_bf16(aq[t], bk[t], c);
#pragma unroll
    for (int j = 0; j < 8; ++j) {
      int m = half * 8 + j;
      int k = kbase + l16;
      sc[m][k] = (k <= q0 + m) ? c[j] * scale : NEG;
    }
  }
  __syncthreads();

  // ---- phase 2: top-128 threshold + softmax (one wave per 4 rows) ----
  for (int r = wave * 4; r < wave * 4 + 4; ++r) {
    unsigned int key[32];
    float        val[32];
    float m = NEG;
#pragma unroll
    for (int i = 0; i < 32; ++i) {
      float v = sc[r][lane + 32 * i];
      val[i] = v;
      unsigned int b = __builtin_bit_cast(unsigned int, v);
      key[i] = b ^ ((b >> 31) ? 0xFFFFFFFFu : 0x80000000u);  // order-preserving map
      m = fmaxf(m, v);
    }
    for (int off = 16; off; off >>= 1) m = fmaxf(m, __shfl_xor(m, off, 32));
    unsigned int thr = 0u;                     // exact 128th-largest key
    for (int bit = 31; bit >= 0; --bit) {
      unsigned int cand = thr | (1u << bit);
      int cnt = 0;
#pragma unroll
      for (int i = 0; i < 32; ++i) cnt += (key[i] >= cand) ? 1 : 0;
      for (int off = 16; off; off >>= 1) cnt += __shfl_xor(cnt, off, 32);
      if (cnt >= TOPK) thr = cand;
    }
    float sum = 0.f;
#pragma unroll
    for (int i = 0; i < 32; ++i) {
      float p = (key[i] >= thr) ? __expf(val[i] - m) : 0.f;
      val[i] = p;
      sum += p;
    }
    for (int off = 16; off; off >>= 1) sum += __shfl_xor(sum, off, 32);
    float inv = 1.f / sum;
    unsigned short* prow = (unsigned short*)&sc[r][0];   // in-place bf16 probs
#pragma unroll
    for (int i = 0; i < 32; ++i) prow[lane + 32 * i] = f2bf(val[i] * inv);
  }
  __syncthreads();

  // ---- phase 3: out = probs @ V (wave owns d columns wave*32 .. +32) ----
  v8f o0 = zero, o1 = zero;
  const unsigned short* vb   = vT + (size_t)kvh * HD * S_LEN;
  const unsigned short* prow = (const unsigned short*)&sc[l16][0];
  for (int k0 = 0; k0 < S_LEN; k0 += 32) {
    v16bf af = ld16(prow + k0 + half * 16);
    const unsigned short* b0 = vb + (size_t)(wave * 32 + l16) * S_LEN + k0 + half * 16;
    v16bf v0 = ld16(b0), v1 = ld16(b0 + 16 * S_LEN);
    o0 = wmma_bf16(af, v0, o0);
    o1 = wmma_bf16(af, v1, o1);
  }
#pragma unroll
  for (int j = 0; j < 8; ++j) {
    int m = half * 8 + j;
    outb[(size_t)(q0 + m) * DMODEL + h * HD + wave * 32 + l16]      = f2bf(o0[j]);
    outb[(size_t)(q0 + m) * DMODEL + h * HD + wave * 32 + 16 + l16] = f2bf(o1[j]);
  }
}

// ---------------------------------------------------------------------------
extern "C" void kernel_launch(void* const* d_in, const int* in_sizes, int n_in,
                              void* d_out, int out_size, void* d_ws, size_t ws_size,
                              hipStream_t stream) {
  (void)in_sizes; (void)n_in; (void)out_size; (void)ws_size;
  const float* X  = (const float*)d_in[0];
  const float* Wq = (const float*)d_in[1];
  const float* Wk = (const float*)d_in[2];
  const float* Wv = (const float*)d_in[3];
  const float* Wo = (const float*)d_in[4];

  char* ws = (char*)d_ws;
  unsigned short* Xb  = (unsigned short*)ws; ws += (size_t)S_LEN * DMODEL * 2;
  unsigned short* Wqb = (unsigned short*)ws; ws += (size_t)DMODEL * DMODEL * 2;
  unsigned short* Wkb = (unsigned short*)ws; ws += (size_t)NKV * HD * DMODEL * 2;
  unsigned short* Wvb = (unsigned short*)ws; ws += (size_t)NKV * HD * DMODEL * 2;
  unsigned short* Wob = (unsigned short*)ws; ws += (size_t)DMODEL * DMODEL * 2;
  float* Qf = (float*)ws;                    ws += (size_t)S_LEN * DMODEL * 4;
  float* Kf = (float*)ws;                    ws += (size_t)S_LEN * NKV * HD * 4;
  float* Vf = (float*)ws;                    ws += (size_t)S_LEN * NKV * HD * 4;
  unsigned short* qrb = (unsigned short*)ws; ws += (size_t)NHEAD * S_LEN * HD * 2;
  unsigned short* krb = (unsigned short*)ws; ws += (size_t)NKV * S_LEN * HD * 2;
  unsigned short* vTb = (unsigned short*)ws; ws += (size_t)NKV * HD * S_LEN * 2;
  unsigned short* attnb = (unsigned short*)ws; ws += (size_t)S_LEN * DMODEL * 2;

  auto cvt = [&](const float* src, unsigned short* dst, size_t n) {
    cvt_bf16<<<(int)((n / 8 + 255) / 256), 256, 0, stream>>>(src, dst, (int)n);
  };
  cvt(X,  Xb,  (size_t)S_LEN * DMODEL);
  cvt(Wq, Wqb, (size_t)DMODEL * DMODEL);
  cvt(Wk, Wkb, (size_t)NKV * HD * DMODEL);
  cvt(Wv, Wvb, (size_t)NKV * HD * DMODEL);
  cvt(Wo, Wob, (size_t)DMODEL * DMODEL);

  gemm_bb<<<dim3(DMODEL / 128, S_LEN / 128), 256, 0, stream>>>(Xb, Wqb, Qf, S_LEN, DMODEL, DMODEL);
  gemm_bb<<<dim3(NKV * HD / 128, S_LEN / 128), 256, 0, stream>>>(Xb, Wkb, Kf, S_LEN, NKV * HD, DMODEL);
  gemm_bb<<<dim3(NKV * HD / 128, S_LEN / 128), 256, 0, stream>>>(Xb, Wvb, Vf, S_LEN, NKV * HD, DMODEL);

  int tot = S_LEN * DMODEL + 2 * S_LEN * NKV * HD;
  rope_pack<<<(tot + 255) / 256, 256, 0, stream>>>(Qf, Kf, Vf, qrb, krb, vTb);

  attn_sparse<<<dim3(S_LEN / 16, NHEAD), 128, 0, stream>>>(qrb, krb, vTb, attnb);

  gemm_bb<<<dim3(DMODEL / 128, S_LEN / 128), 256, 0, stream>>>(attnb, Wob, (float*)d_out, S_LEN, DMODEL, DMODEL);
}
